// TreeLSTM_26809185862284
// MI455X (gfx1250) — matrix operand order repaired and checked by reading, not compiled, not run on hardware
//
#include <hip/hip_runtime.h>
#include <hip/hip_bf16.h>
#include <math.h>

// ---------------- constants ----------------
#define NN      32767      // nodes
#define INC     512        // in_channels
#define HD      256        // hidden
#define G4      1024       // 4*H
#define DEPTH   15

typedef _Float16 half_t;
typedef __attribute__((ext_vector_type(8)))  _Float16 v8h;
typedef __attribute__((ext_vector_type(16))) _Float16 v16h;
typedef __attribute__((ext_vector_type(8)))  float    v8f;

__device__ __forceinline__ v16h join8(v8h a0, v8h a1) {
    v16h r;
#pragma unroll
    for (int j = 0; j < 8; ++j) { r[j] = a0[j]; r[j + 8] = a1[j]; }
    return r;
}

__device__ __forceinline__ float sigm(float x) { return 1.0f / (1.0f + __expf(-x)); }

// ---------------- conversion / packing ----------------

// x (f32 [N,512]) -> f16 copy
__global__ __launch_bounds__(256) void k_cvt_x(const float* __restrict__ x,
                                               half_t* __restrict__ x16, int total) {
    int i = blockIdx.x * blockDim.x + threadIdx.x;
    if (i < total) x16[i] = (half_t)x[i];
}

// Pack W [1024, K] (row major, gates x K) into per-lane WMMA B layout:
// Bp[ ((gt*KB + kbi)*32 + lane)*16 + j ] = W[gcol*K + kbi*32 + (lane>>4)*8 + (j&7) + (j>>3)*16],
// gcol = gt*16 + (lane&15).  One contiguous 32B load per lane per K-step.
__global__ __launch_bounds__(256) void k_pack_w(const float* __restrict__ W,
                                                half_t* __restrict__ Bp, int K) {
    int idx = blockIdx.x * blockDim.x + threadIdx.x;
    int total = G4 * K;
    if (idx >= total) return;
    int j    = idx & 15;
    int lane = (idx >> 4) & 31;
    int rest = idx >> 9;          // gt*KB + kbi
    int KB   = K >> 5;
    int kbi  = rest % KB;
    int gt   = rest / KB;
    int gcol = gt * 16 + (lane & 15);
    int k    = kbi * 32 + (lane >> 4) * 8 + (j & 7) + (j >> 3) * 16;
    Bp[idx]  = (half_t)W[gcol * K + k];
}

// ---------------- big GEMM: xpre = x @ W_ih^T + b_ih + b_hh ----------------
// One wave = 16x64 output strip (4 gate tiles share one A fragment -> 4x the
// WMMA work per A byte).  M tiles = 2048, gate quads = 16, K = 512 (16 steps,
// 4 v_wmma per step).  Result stored as f16 (halves the dominant HBM stream).
__global__ __launch_bounds__(256) void k_gemm_xpre(const half_t* __restrict__ x16,
                                                   const half_t* __restrict__ Bp,
                                                   const float* __restrict__ b_ih,
                                                   const float* __restrict__ b_hh,
                                                   half_t* __restrict__ xpre) {
    const int MT = (NN + 15) >> 4;                         // 2048
    const int KB = INC / 32;                               // 16
    int wave = blockIdx.x * (blockDim.x >> 5) + (threadIdx.x >> 5);
    int mt  = wave >> 4;                                   // /16 gate-quads
    int gq  = wave & 15;                                   // gate quad: tiles 4*gq..4*gq+3
    if (mt >= MT) return;

    int lane = threadIdx.x & 31;
    int m  = lane & 15;
    int hs = lane >> 4;
    int row = mt * 16 + m;
    int rowc = row < NN ? row : (NN - 1);                  // clamp (last tile)

    v8f acc0 = {}, acc1 = {}, acc2 = {}, acc3 = {};
    const half_t* bbase = Bp + (size_t)(4 * gq) * KB * 32 * 16;
    const size_t bstride = (size_t)KB * 32 * 16;           // one gate tile of packed B
#pragma unroll
    for (int kbi = 0; kbi < KB; ++kbi) {
        const half_t* ap = x16 + (size_t)rowc * INC + kbi * 32 + hs * 8;
        v8h a0 = *(const v8h*)ap;
        v8h a1 = *(const v8h*)(ap + 16);
        v16h A = join8(a0, a1);
        size_t boff = (size_t)(kbi * 32 + lane) * 16;
        v16h B0 = *(const v16h*)(bbase + 0 * bstride + boff);
        v16h B1 = *(const v16h*)(bbase + 1 * bstride + boff);
        v16h B2 = *(const v16h*)(bbase + 2 * bstride + boff);
        v16h B3 = *(const v16h*)(bbase + 3 * bstride + boff);
        acc0 = __builtin_amdgcn_wmma_f32_16x16x32_f16(false, A, false, B0, (short)0, acc0, false, false);
        acc1 = __builtin_amdgcn_wmma_f32_16x16x32_f16(false, A, false, B1, (short)0, acc1, false, false);
        acc2 = __builtin_amdgcn_wmma_f32_16x16x32_f16(false, A, false, B2, (short)0, acc2, false, false);
        acc3 = __builtin_amdgcn_wmma_f32_16x16x32_f16(false, A, false, B3, (short)0, acc3, false, false);
    }

    int n = lane & 15;
#pragma unroll
    for (int q = 0; q < 4; ++q) {
        const v8f& acc = (q == 0) ? acc0 : (q == 1) ? acc1 : (q == 2) ? acc2 : acc3;
        int gcol = (4 * gq + q) * 16 + n;
        float bias = b_ih[gcol] + b_hh[gcol];
#pragma unroll
        for (int v = 0; v < 8; ++v) {
            int r = mt * 16 + v + hs * 8;
            if (r < NN) xpre[(size_t)r * G4 + gcol] = (half_t)(acc[v] + bias);
        }
    }
}

// ---------------- per-level prep: mean of children h (f16) and c (f32) ----------------
__global__ __launch_bounds__(256) void k_level_prep(const float* __restrict__ h,
                                                    const float* __restrict__ c,
                                                    half_t* __restrict__ mh16,
                                                    float* __restrict__ mc,
                                                    int lo, int n_level) {
    int idx = blockIdx.x * blockDim.x + threadIdx.x;
    if (idx >= n_level * HD) return;
    int local = idx >> 8;
    int hidx  = idx & 255;
    int node  = lo + local;
    int c1 = 2 * node + 1;
    float mh = 0.0f, mcc = 0.0f;
    if (c1 < NN) {                                         // complete tree: both children
        mh  = 0.5f * (h[(size_t)c1 * HD + hidx] + h[(size_t)(c1 + 1) * HD + hidx]);
        mcc = 0.5f * (c[(size_t)c1 * HD + hidx] + c[(size_t)(c1 + 1) * HD + hidx]);
    }
    mh16[idx] = (half_t)mh;
    mc[idx]   = mcc;
}

// ---------------- per-level fused GEMM + LSTM cell ----------------
// one wave owns a 16-node x 16-hidden tile, carrying FOUR accumulators
// (i, f, g, o gate blocks at hidden-tile ht, 16+ht, 32+ht, 48+ht) so the
// gate coupling is resolved in registers. 4 v_wmma per K-step.
__global__ __launch_bounds__(256) void k_level_lstm(const half_t* __restrict__ mh16,
                                                    const half_t* __restrict__ Bhh,
                                                    const float* __restrict__ mc,
                                                    const half_t* __restrict__ xpre,
                                                    float* __restrict__ h,
                                                    float* __restrict__ c,
                                                    int lo, int n_level) {
    int mtiles = (n_level + 15) >> 4;
    int wave = blockIdx.x * (blockDim.x >> 5) + (threadIdx.x >> 5);
    int mt = wave >> 4;                                    // /16 hidden tiles
    int ht = wave & 15;
    if (mt >= mtiles) return;

    int lane = threadIdx.x & 31;
    int m  = lane & 15;
    int hs = lane >> 4;
    int lrow = mt * 16 + m;
    if (lrow >= n_level) lrow = n_level - 1;               // clamp for partial tiles

    v8f ai = {}, af = {}, ag = {}, ao = {};
    const int KB = HD / 32;                                // 8
#pragma unroll
    for (int kbi = 0; kbi < KB; ++kbi) {
        const half_t* ap = mh16 + (size_t)lrow * HD + kbi * 32 + hs * 8;
        v8h a0 = *(const v8h*)ap;
        v8h a1 = *(const v8h*)(ap + 16);
        v16h A = join8(a0, a1);
        v16h Bi = *(const v16h*)(Bhh + (size_t)((((ht     ) * KB + kbi) * 32 + lane)) * 16);
        v16h Bf = *(const v16h*)(Bhh + (size_t)((((16 + ht) * KB + kbi) * 32 + lane)) * 16);
        v16h Bg = *(const v16h*)(Bhh + (size_t)((((32 + ht) * KB + kbi) * 32 + lane)) * 16);
        v16h Bo = *(const v16h*)(Bhh + (size_t)((((48 + ht) * KB + kbi) * 32 + lane)) * 16);
        ai = __builtin_amdgcn_wmma_f32_16x16x32_f16(false, A, false, Bi, (short)0, ai, false, false);
        af = __builtin_amdgcn_wmma_f32_16x16x32_f16(false, A, false, Bf, (short)0, af, false, false);
        ag = __builtin_amdgcn_wmma_f32_16x16x32_f16(false, A, false, Bg, (short)0, ag, false, false);
        ao = __builtin_amdgcn_wmma_f32_16x16x32_f16(false, A, false, Bo, (short)0, ao, false, false);
    }

    int hidx = ht * 16 + (lane & 15);
#pragma unroll
    for (int v = 0; v < 8; ++v) {
        int local = mt * 16 + v + hs * 8;
        if (local < n_level) {
            int node = lo + local;
            const half_t* xp = xpre + (size_t)node * G4;
            float gi = ai[v] + (float)xp[hidx];
            float gf = af[v] + (float)xp[256 + hidx];
            float gg = ag[v] + (float)xp[512 + hidx];
            float go = ao[v] + (float)xp[768 + hidx];
            float mcc = mc[(size_t)local * HD + hidx];
            float cn = sigm(gf) * mcc + sigm(gi) * tanhf(gg);
            float hn = sigm(go) * tanhf(cn);
            c[(size_t)node * HD + hidx] = cn;
            h[(size_t)node * HD + hidx] = hn;
        }
    }
}

// ---------------- pooled mean + fc ----------------
__global__ __launch_bounds__(256) void k_pool_fc(const float* __restrict__ h,
                                                 const float* __restrict__ W_fc,
                                                 const float* __restrict__ b_fc,
                                                 float* __restrict__ out) {
    __shared__ float red[256];
    int t = threadIdx.x;
    float acc = 0.0f;
    const long total = (long)NN * HD;
    for (long idx = t; idx < total; idx += 256) {
        acc += h[idx] * W_fc[(int)(idx & 255)];
    }
    red[t] = acc;
    __syncthreads();
    for (int s = 128; s > 0; s >>= 1) {
        if (t < s) red[t] += red[t + s];
        __syncthreads();
    }
    if (t == 0) out[0] = red[0] / (float)NN + b_fc[0];
}

// ---------------- launcher ----------------
static inline size_t align256(size_t x) { return (x + 255) & ~(size_t)255; }

extern "C" void kernel_launch(void* const* d_in, const int* in_sizes, int n_in,
                              void* d_out, int out_size, void* d_ws, size_t ws_size,
                              hipStream_t stream) {
    const float* x     = (const float*)d_in[0];
    // d_in[1] = edge_index (unused: complete binary tree, children of i are 2i+1, 2i+2)
    const float* W_ih  = (const float*)d_in[2];
    const float* W_hh  = (const float*)d_in[3];
    const float* b_ih  = (const float*)d_in[4];
    const float* b_hh  = (const float*)d_in[5];
    const float* W_fc  = (const float*)d_in[6];
    const float* b_fc  = (const float*)d_in[7];
    float* out = (float*)d_out;

    // workspace carve-up
    char* ws = (char*)d_ws;
    size_t off = 0;
    half_t* x16  = (half_t*)(ws + off); off = align256(off + (size_t)NN * INC * sizeof(half_t));
    half_t* Bih  = (half_t*)(ws + off); off = align256(off + (size_t)G4 * INC * sizeof(half_t));
    half_t* Bhh  = (half_t*)(ws + off); off = align256(off + (size_t)G4 * HD * sizeof(half_t));
    half_t* xpre = (half_t*)(ws + off); off = align256(off + (size_t)NN * G4 * sizeof(half_t));
    float*  hbuf = (float*) (ws + off); off = align256(off + (size_t)NN * HD * sizeof(float));
    float*  cbuf = (float*) (ws + off); off = align256(off + (size_t)NN * HD * sizeof(float));
    half_t* mh16 = (half_t*)(ws + off); off = align256(off + (size_t)16384 * HD * sizeof(half_t));
    float*  mc   = (float*) (ws + off); off = align256(off + (size_t)16384 * HD * sizeof(float));

    // 1) f16 conversion + weight packing
    {
        int total = NN * INC;
        k_cvt_x<<<(total + 255) / 256, 256, 0, stream>>>(x, x16, total);
        k_pack_w<<<(G4 * INC + 255) / 256, 256, 0, stream>>>(W_ih, Bih, INC);
        k_pack_w<<<(G4 * HD + 255) / 256, 256, 0, stream>>>(W_hh, Bhh, HD);
    }

    // 2) hoisted input projection: xpre = x @ W_ih^T + b_ih + b_hh   (WMMA)
    {
        int mtiles = (NN + 15) / 16;            // 2048
        int waves  = mtiles * 16;               // 16 gate-quads (4 tiles each)
        k_gemm_xpre<<<(waves + 7) / 8, 256, 0, stream>>>(x16, Bih, b_ih, b_hh, xpre);
    }

    // 3) bottom-up level sweep (children written before parents read them)
    for (int d = DEPTH - 1; d >= 0; --d) {
        int lo = (1 << d) - 1;
        int hi = (1 << (d + 1)) - 1;
        if (hi > NN) hi = NN;
        int n_level = hi - lo;

        int prep_threads = n_level * HD;
        k_level_prep<<<(prep_threads + 255) / 256, 256, 0, stream>>>(
            hbuf, cbuf, mh16, mc, lo, n_level);

        int mtiles = (n_level + 15) / 16;
        int waves  = mtiles * 16;               // 16 hidden tiles per node tile
        k_level_lstm<<<(waves + 7) / 8, 256, 0, stream>>>(
            mh16, Bhh, mc, xpre, hbuf, cbuf, lo, n_level);
    }

    // 4) pooled mean over all h, then fc -> scalar
    k_pool_fc<<<1, 256, 0, stream>>>(hbuf, W_fc, b_fc, out);
    (void)in_sizes; (void)n_in; (void)out_size; (void)ws_size;
}
